// RNN_79250736546387
// MI455X (gfx1250) — matrix operand order repaired and checked by reading, not compiled, not run on hardware
//
#include <hip/hip_runtime.h>

typedef __attribute__((ext_vector_type(2)))  float  v2f;
typedef __attribute__((ext_vector_type(8)))  float  v8f;
typedef __attribute__((ext_vector_type(8)))  __bf16 v8bf;
typedef __attribute__((ext_vector_type(16))) __bf16 v16bf;
typedef __attribute__((ext_vector_type(4)))  unsigned int u32x4;
typedef __attribute__((ext_vector_type(8)))  int i32x8;
typedef __attribute__((ext_vector_type(4)))  int i32x4;

#define B_  64
#define T_  2048
#define I_  128
#define H_  256
#define O_  128

#if defined(__has_builtin)
#if __has_builtin(__builtin_amdgcn_tensor_load_to_lds)
#define HAVE_TDM 1
#endif
#endif

// LDS byte-offset map for the scan kernel
#define LDS_WHH   0                         // bf16 [16nt][8kb][32lane][16] = 128 KB
#define LDS_WCH   (H_ * H_ * 2)             // bf16 [8nt][8kb][32lane][16] =  64 KB
#define LDS_HS    (LDS_WCH + H_ * O_ * 2)   // bf16 [16][256]              =   8 KB
#define LDS_XH    (LDS_HS + 16 * H_ * 2)    // f32 3 x [16][256]           =  48 KB
#define LDS_XC    (LDS_XH + 3 * 16 * H_ * 4)// f32 3 x [16][128]           =  24 KB
#if HAVE_TDM
#define LDS_BYTES (LDS_XC + 3 * 16 * O_ * 4)
#else
#define LDS_BYTES LDS_XH
#endif

__device__ __forceinline__ __bf16 f2bf(float f) {
  unsigned u = __builtin_bit_cast(unsigned, f);
  u += 0x7fffu + ((u >> 16) & 1u);           // round-to-nearest-even
  unsigned short s = (unsigned short)(u >> 16);
  return __builtin_bit_cast(__bf16, s);
}

// ---------------------------------------------------------------------------
// Kernel 1: Y[M,N] = X[M,128] @ W[128,N] + bias, fp32 WMMA 16x16x4.
// One wave per 16x16 tile; W tile staged once per block in LDS in B-fragment
// order so the hot loop is 1 global b64 + 1 ds b64 per WMMA.
// ---------------------------------------------------------------------------
__global__ void proj_gemm_f32(const float* __restrict__ X,
                              const float* __restrict__ W,
                              const float* __restrict__ bias,
                              float* __restrict__ out, int N) {
  __shared__ float Wl[32 * 64];              // 8 KB: [kstep][lane][2]
  const int lane = threadIdx.x;              // 0..31
  const int lo = lane & 15, hi = lane >> 4;
  const int nt = blockIdx.x;
  const int n0 = nt * 16;
  const int tid = threadIdx.y * 32 + lane;   // 0..127

  // Stage W tile: thread r owns K-row r; scatter its 16 cols into frag order.
  {
    const float* wrow = W + (size_t)tid * N + n0;
    const int ks = tid >> 2, rhi = (tid >> 1) & 1, j = tid & 1;
    for (int c = 0; c < 16; ++c)
      Wl[ks * 64 + (rhi * 16 + c) * 2 + j] = wrow[c];
  }
  __syncthreads();

  const int mt = blockIdx.y * 4 + threadIdx.y;
  const size_t m0 = (size_t)mt * 16;
  const float* arow = X + (m0 + lo) * I_ + 2 * hi;

  v8f acc = {};
#pragma unroll
  for (int ks = 0; ks < 32; ++ks) {
    v2f a = *(const v2f*)(arow + 4 * ks);            // K = 4ks+2hi, +1
    v2f b = *(const v2f*)(&Wl[ks * 64 + lane * 2]);  // same K, col n0+lo
    acc = __builtin_amdgcn_wmma_f32_16x16x4_f32(
        false, a, false, b, (short)0, acc, false, false);
  }
  const float bias_n = bias[n0 + lo];
  for (int v = 0; v < 8; ++v)                        // C row = v + 8*hi
    out[(m0 + v + 8 * hi) * N + n0 + lo] = acc[v] + bias_n;
}

// ---------------------------------------------------------------------------
// TDM: load a 16 x width fp32 tile (row stride = stride elems) into LDS.
// D# per CDNA5 ISA. Wave-uniform args; issued by wave 0.
// ---------------------------------------------------------------------------
#if HAVE_TDM
__device__ __forceinline__ void tdm_load_tile(unsigned lds_off,
                                              const float* gptr,
                                              unsigned width,
                                              unsigned stride) {
  unsigned long long ga = (unsigned long long)(const void*)gptr;
  u32x4 g0;
  g0[0] = 1u;                                        // count=1, user D#
  g0[1] = lds_off;                                   // lds_addr (bytes)
  g0[2] = (unsigned)(ga & 0xffffffffu);              // global_addr[31:0]
  g0[3] = (unsigned)((ga >> 32) & 0x01ffffffu) | (2u << 30); // [56:32] | type=2
  i32x8 g1;
  g1[0] = (int)(2u << 16);                           // data_size=4B, mask=0
  g1[1] = (int)((width & 0xffffu) << 16);            // tensor_dim0[15:0]
  g1[2] = (int)(16u << 16);                          // dim0 hi=0 | tensor_dim1=16
  g1[3] = (int)((width & 0xffffu) << 16);            // dim1 hi=0 | tile_dim0
  g1[4] = 16;                                        // tile_dim1=16, tile_dim2=0
  g1[5] = (int)stride;                               // tensor_dim0_stride[31:0]
  g1[6] = 0;                                         // stride hi | dim1_stride lo
  g1[7] = 0;
  i32x4 z4 = {};
  i32x8 z8 = {};
  __builtin_amdgcn_tensor_load_to_lds(g0, g1, z4, z4, z8, 0);
}
#endif

// ---------------------------------------------------------------------------
// Kernel 2: sequential scan over T. One block per 16-row batch tile,
// 512 threads = 16 waves; wave nt owns h columns [nt*16, nt*16+16).
// Whh/Wch cached in LDS (bf16, B-fragment packed). xh/xc tiles triple-buffered
// in LDS via the Tensor Data Mover, issued 2 steps ahead by wave 0.
// hOut holds xh on entry and is overwritten in place with h (same cOut/xc).
// ---------------------------------------------------------------------------
__global__ void rnn_scan(const float* __restrict__ Whh,
                         const float* __restrict__ Wch,
                         float* __restrict__ cOut,
                         float* __restrict__ hOut) {
  extern __shared__ __align__(16) char smem[];
  __bf16* WhhP = (__bf16*)(smem + LDS_WHH);
  __bf16* WchP = (__bf16*)(smem + LDS_WCH);
  __bf16* hS   = (__bf16*)(smem + LDS_HS);
#if HAVE_TDM
  float*  xhBuf = (float*)(smem + LDS_XH);   // 3 slots of [16][256]
  float*  xcBuf = (float*)(smem + LDS_XC);   // 3 slots of [16][128]
#endif

  const int tid  = threadIdx.x;
  const int lane = tid & 31;
  const int wave = tid >> 5;                 // 0..15 == N tile
  const int lo = lane & 15, hi = lane >> 4;
  const int nt = wave;
  const int mb = blockIdx.x;                 // batch tile: rows mb*16..+15

  // ---- pack weights into bf16 B-fragment order ----
  for (int idx = tid; idx < H_ * H_; idx += blockDim.x) {
    int j = idx & 15, ln = (idx >> 4) & 31, kb = (idx >> 9) & 7, nn = idx >> 12;
    int k = kb * 32 + (j >> 3) * 16 + (ln >> 4) * 8 + (j & 7);
    int n = nn * 16 + (ln & 15);
    WhhP[idx] = f2bf(Whh[k * H_ + n]);
  }
  for (int idx = tid; idx < H_ * O_; idx += blockDim.x) {
    int j = idx & 15, ln = (idx >> 4) & 31, kb = (idx >> 9) & 7, nn = idx >> 12;
    int k = kb * 32 + (j >> 3) * 16 + (ln >> 4) * 8 + (j & 7);
    int n = nn * 16 + (ln & 15);
    WchP[idx] = f2bf(Wch[k * O_ + n]);
  }
  for (int idx = tid; idx < 16 * H_; idx += blockDim.x) hS[idx] = f2bf(0.f);

  const float* xhG = hOut + (size_t)(mb * 16) * T_ * H_;  // tile base, +t*H_
  const float* xcG = cOut + (size_t)(mb * 16) * T_ * O_;  // tile base, +t*O_

#if HAVE_TDM
  if (wave == 0) {                           // prefill slots 0 and 1
    tdm_load_tile(LDS_XH,          xhG,      H_, T_ * H_);
    tdm_load_tile(LDS_XC,          xcG,      O_, T_ * O_);
    tdm_load_tile(LDS_XH + 16384,  xhG + H_, H_, T_ * H_);
    tdm_load_tile(LDS_XC + 8192,   xcG + O_, O_, T_ * O_);
    __builtin_amdgcn_s_wait_tensorcnt(2);    // slot 0 complete (in-order)
  }
#endif
  __syncthreads();

  // per-lane output element bases (C layout: row = v + 8*hi, col = nt*16+lo)
  size_t hBase[8], cBase[8];
  for (int v = 0; v < 8; ++v) {
    size_t b = (size_t)(mb * 16 + v + 8 * hi);
    hBase[v] = b * T_ * H_ + nt * 16 + lo;
    cBase[v] = b * T_ * O_ + nt * 16 + lo;
  }

  for (int t = 0; t < T_; ++t) {
#if HAVE_TDM
    const int slot = t % 3;
    const float* xhT = xhBuf + slot * (16 * H_);
    const float* xcT = xcBuf + slot * (16 * O_);
#else
    float xh[8];
    for (int v = 0; v < 8; ++v) xh[v] = hOut[hBase[v] + (size_t)t * H_];
    float xc[8];
    if (nt < 8)
      for (int v = 0; v < 8; ++v) xc[v] = cOut[cBase[v] + (size_t)t * O_];
    if (t + 1 < T_)
      for (int v = 0; v < 8; ++v)
        __builtin_prefetch(&hOut[hBase[v] + (size_t)(t + 1) * H_], 0, 0);
#endif

    // ---- h_new_tile = h @ Whh[:, nt*16 : nt*16+16] ----
    v8f acc = {};
    const __bf16* wb = WhhP + (size_t)nt * (8 * 512);
#pragma unroll
    for (int kb = 0; kb < 8; ++kb) {
      const __bf16* ap = hS + lo * H_ + kb * 32 + hi * 8;
      v8bf a0 = *(const v8bf*)(ap);          // K = k0+hi*8+{0..7}
      v8bf a1 = *(const v8bf*)(ap + 16);     // K = k0+16+hi*8+{0..7}
      v16bf a = __builtin_shufflevector(a0, a1, 0, 1, 2, 3, 4, 5, 6, 7,
                                        8, 9, 10, 11, 12, 13, 14, 15);
      v16bf b = *(const v16bf*)(wb + kb * 512 + lane * 16);
      acc = __builtin_amdgcn_wmma_f32_16x16x32_bf16(
          false, a, false, b, (short)0, acc, false, false);
    }
    float hn[8];
    for (int v = 0; v < 8; ++v) {
#if HAVE_TDM
      float xh_v = xhT[(v + 8 * hi) * H_ + nt * 16 + lo];
#else
      float xh_v = xh[v];
#endif
      hn[v] = tanhf(acc[v] + xh_v);
    }

    __syncthreads();                         // all waves done reading old hS
    for (int v = 0; v < 8; ++v) {
      hS[(v + 8 * hi) * H_ + nt * 16 + lo] = f2bf(hn[v]);
      hOut[hBase[v] + (size_t)t * H_] = hn[v];     // overwrite xh with h
    }

#if HAVE_TDM
    if (wave == 0) {                         // fetch t+2; guarantee t+1 ready
      if (t + 2 < T_) {
        const int s2 = (t + 2) % 3;
        tdm_load_tile(LDS_XH + s2 * 16384, xhG + (size_t)(t + 2) * H_, H_, T_ * H_);
        tdm_load_tile(LDS_XC + s2 * 8192,  xcG + (size_t)(t + 2) * O_, O_, T_ * O_);
        __builtin_amdgcn_s_wait_tensorcnt(2);
      } else if (t + 1 < T_) {
        __builtin_amdgcn_s_wait_tensorcnt(0);
      }
    }
#endif
    __syncthreads();                         // new hS (and slot t+1) visible

    // ---- c_tile = xc_t + h_new @ Wch[:, nt*16 : nt*16+16]  (waves 0..7) ----
    if (nt < 8) {
      v8f acc2 = {};
      const __bf16* wc = WchP + (size_t)nt * (8 * 512);
#pragma unroll
      for (int kb = 0; kb < 8; ++kb) {
        const __bf16* ap = hS + lo * H_ + kb * 32 + hi * 8;
        v8bf a0 = *(const v8bf*)(ap);
        v8bf a1 = *(const v8bf*)(ap + 16);
        v16bf a = __builtin_shufflevector(a0, a1, 0, 1, 2, 3, 4, 5, 6, 7,
                                          8, 9, 10, 11, 12, 13, 14, 15);
        v16bf b = *(const v16bf*)(wc + kb * 512 + lane * 16);
        acc2 = __builtin_amdgcn_wmma_f32_16x16x32_bf16(
            false, a, false, b, (short)0, acc2, false, false);
      }
      for (int v = 0; v < 8; ++v) {
#if HAVE_TDM
        float xc_v = xcT[(v + 8 * hi) * O_ + nt * 16 + lo];
#else
        float xc_v = xc[v];
#endif
        cOut[cBase[v] + (size_t)t * O_] = acc2[v] + xc_v;   // overwrite xc
      }
    }
  }
}

// ---------------------------------------------------------------------------
extern "C" void kernel_launch(void* const* d_in, const int* in_sizes, int n_in,
                              void* d_out, int out_size, void* d_ws, size_t ws_size,
                              hipStream_t stream) {
  (void)in_sizes; (void)n_in; (void)out_size; (void)d_ws; (void)ws_size;
  const float* x   = (const float*)d_in[0];
  const float* Whx = (const float*)d_in[1];
  const float* Whh = (const float*)d_in[2];
  const float* bh  = (const float*)d_in[3];
  const float* Wch = (const float*)d_in[4];
  const float* Wcx = (const float*)d_in[5];
  const float* bc  = (const float*)d_in[6];

  float* cOut = (float*)d_out;                       // [B,T,O]
  float* hOut = cOut + (size_t)B_ * T_ * O_;         // [B,T,H]

  const int MT = (B_ * T_) / 16;                     // 8192 M tiles
  dim3 blk(32, 4);                                   // 4 waves / block
  proj_gemm_f32<<<dim3(H_ / 16, MT / 4), blk, 0, stream>>>(x, Whx, bh, hOut, H_);
  proj_gemm_f32<<<dim3(O_ / 16, MT / 4), blk, 0, stream>>>(x, Wcx, bc, cOut, O_);

  rnn_scan<<<4, 512, (size_t)LDS_BYTES, stream>>>(Whh, Wch, cOut, hOut);
}
